// Attention_55654186222026
// MI455X (gfx1250) — matrix operand order repaired and checked by reading, not compiled, not run on hardware
//
#include <hip/hip_runtime.h>
#include <hip/hip_bf16.h>
#include <math.h>

// CDNA5 (gfx1250) wave32 WMMA types
typedef __attribute__((ext_vector_type(16))) __bf16 v16bf;
typedef __attribute__((ext_vector_type(8)))  float  v8f;

#define BDIM 32
#define TDIM 2048
#define HDIM 1024
#define TC   128                 // t-rows per block (8 waves x 16 rows)
#define NCHUNK (TDIM / TC)       // 16 chunks per batch

__device__ __forceinline__ __bf16 f2bf(float f) { return (__bf16)f; }

// ---------------------------------------------------------------------------
// Kernel A: u[h] = sum_k W_attn[k, H + h] * W_v[k]   (We^T @ W_v), store bf16
// ---------------------------------------------------------------------------
__global__ __launch_bounds__(256)
void attn_prep_u(const float* __restrict__ Wattn, const float* __restrict__ Wv,
                 __bf16* __restrict__ ubf) {
  const int h = blockIdx.x * 256 + threadIdx.x;
  float acc = 0.f;
#pragma unroll 8
  for (int k = 0; k < HDIM; ++k)
    acc = fmaf(Wattn[(size_t)k * (2 * HDIM) + HDIM + h], Wv[k], acc);
  ubf[h] = f2bf(acc);
}

// ---------------------------------------------------------------------------
// Kernel B: fused (scores via WMMA bf16) -> chunk softmax -> partial context.
// One block = one (batch b, t-chunk) pair, 256 threads = 8 waves.
// Each wave computes 16 scores: D = A(16x32 enc tile, bf16) x B(u broadcast).
// ---------------------------------------------------------------------------
__global__ __launch_bounds__(256)
void attn_partial(const float* __restrict__ enc, const __bf16* __restrict__ ubf,
                  float* __restrict__ Mp, float* __restrict__ Sp,
                  float* __restrict__ ctxp) {
  __shared__ float s_lds[TC];    // scores, then softmax weights
  __shared__ float red[256];     // reduction scratch

  const int b   = blockIdx.y;
  const int ck  = blockIdx.x;
  const int t0  = ck * TC;
  const int tid = threadIdx.x;
  const int lane = tid & 31;
  const int wv   = tid >> 5;     // wave id 0..7
  const int m    = lane & 15;    // row within 16-row tile
  const int hf   = lane >> 4;    // half-wave selector

  // ---- Phase 1: scores for this wave's 16 rows via v_wmma_f32_16x16x32_bf16
  const float* rowp = enc + ((size_t)b * TDIM + t0 + wv * 16 + m) * HDIM;
  v8f c = {0.f, 0.f, 0.f, 0.f, 0.f, 0.f, 0.f, 0.f};

#pragma unroll 4
  for (int kc = 0; kc < HDIM / 32; ++kc) {
    const int k0 = kc * 32;
    // A layout (16-bit A 16x32): this lane holds K = k0+16*hf..+15 (contiguous)
    const float4 a0 = *(const float4*)(rowp + k0 + 16 * hf);
    const float4 a1 = *(const float4*)(rowp + k0 + 16 * hf + 4);
    const float4 a2 = *(const float4*)(rowp + k0 + 16 * hf + 8);
    const float4 a3 = *(const float4*)(rowp + k0 + 16 * hf + 12);
    v16bf A;
    A[0]  = f2bf(a0.x); A[1]  = f2bf(a0.y); A[2]  = f2bf(a0.z); A[3]  = f2bf(a0.w);
    A[4]  = f2bf(a1.x); A[5]  = f2bf(a1.y); A[6]  = f2bf(a1.z); A[7]  = f2bf(a1.w);
    A[8]  = f2bf(a2.x); A[9]  = f2bf(a2.y); A[10] = f2bf(a2.z); A[11] = f2bf(a2.w);
    A[12] = f2bf(a3.x); A[13] = f2bf(a3.y); A[14] = f2bf(a3.z); A[15] = f2bf(a3.w);
    // B layout (32x16 bf16): lane-uniform per half: element e = u[k0 + 16*hf + e]
    // -> u broadcast across all 16 N columns; every D column holds the scores.
    v16bf Bv = *(const v16bf*)(ubf + k0 + 16 * hf);
    c = __builtin_amdgcn_wmma_f32_16x16x32_bf16(
        /*neg_a=*/false, A, /*neg_b=*/false, Bv,
        /*c_mod=*/(short)0, c, /*reuse_a=*/false, /*reuse_b=*/false);
  }
  // C/D f32 layout: VGPR r holds M = r + 8*hf (all N identical) -> lanes 0/16 write
  if (m == 0) {
#pragma unroll
    for (int r = 0; r < 8; ++r) s_lds[wv * 16 + 8 * hf + r] = c[r];
  }
  __syncthreads();

  // ---- Phase 2: chunk-local softmax stats (max, exp, sum)
  red[tid] = (tid < TC) ? s_lds[tid] : -INFINITY;
  __syncthreads();
  for (int s = 128; s >= 1; s >>= 1) {
    if (tid < s) red[tid] = fmaxf(red[tid], red[tid + s]);
    __syncthreads();
  }
  const float Mc = red[0];
  __syncthreads();
  float w = 0.f;
  if (tid < TC) w = expf(s_lds[tid] - Mc);
  red[tid] = w;
  __syncthreads();
  for (int s = 128; s >= 1; s >>= 1) {
    if (tid < s) red[tid] += red[tid + s];
    __syncthreads();
  }
  const float Sc = red[0];
  __syncthreads();
  if (tid < TC) s_lds[tid] = w;   // unnormalized weights
  if (tid == 0) {
    Mp[b * NCHUNK + ck] = Mc;
    Sp[b * NCHUNK + ck] = Sc;
  }
  __syncthreads();

  // ---- Phase 3: partial context ctx_c[h] = sum_j w_j * enc[t0+j, h]
  // (chunk rows just streamed -> WGP$/L2 hits, no extra HBM traffic)
  const float* cb = enc + ((size_t)b * TDIM + t0) * HDIM;
  float* cp = ctxp + ((size_t)(b * NCHUNK) + ck) * HDIM;
#pragma unroll
  for (int q = 0; q < 4; ++q) {
    const int h = tid + q * 256;
    float acc = 0.f;
#pragma unroll 8
    for (int j = 0; j < TC; ++j)
      acc = fmaf(s_lds[j], cb[(size_t)j * HDIM + h], acc);
    cp[h] = acc;
  }
}

// ---------------------------------------------------------------------------
// Kernel C: cross-chunk flash-softmax rescale + combine. One block per batch.
// ---------------------------------------------------------------------------
__global__ __launch_bounds__(256)
void attn_reduce(const float* __restrict__ Mp, const float* __restrict__ Sp,
                 const float* __restrict__ ctxp, float* __restrict__ out) {
  const int b   = blockIdx.x;
  const int tid = threadIdx.x;

  float M = -INFINITY;
#pragma unroll
  for (int c = 0; c < NCHUNK; ++c) M = fmaxf(M, Mp[b * NCHUNK + c]);
  float sc[NCHUNK];
  float S = 0.f;
#pragma unroll
  for (int c = 0; c < NCHUNK; ++c) {
    sc[c] = expf(Mp[b * NCHUNK + c] - M);
    S = fmaf(Sp[b * NCHUNK + c], sc[c], S);
  }
  const float inv = 1.f / S;

#pragma unroll
  for (int q = 0; q < 4; ++q) {
    const int h = tid + q * 256;
    float acc = 0.f;
#pragma unroll
    for (int c = 0; c < NCHUNK; ++c)
      acc = fmaf(ctxp[((size_t)(b * NCHUNK) + c) * HDIM + h], sc[c], acc);
    out[(size_t)b * HDIM + h] = acc * inv;
  }
}

// ---------------------------------------------------------------------------
extern "C" void kernel_launch(void* const* d_in, const int* in_sizes, int n_in,
                              void* d_out, int out_size, void* d_ws, size_t ws_size,
                              hipStream_t stream) {
  (void)in_sizes; (void)n_in; (void)out_size; (void)ws_size;
  // setup_inputs order: encoder_outputs, hidden, W_attn, b_attn, W_v, b_v
  const float* enc   = (const float*)d_in[0];
  const float* Wattn = (const float*)d_in[2];
  const float* Wv    = (const float*)d_in[4];
  float* out = (float*)d_out;

  char* ws = (char*)d_ws;
  __bf16* ubf = (__bf16*)ws;                      // 1024 * 2 B
  float*  Mp  = (float*)(ws + 4096);              // 32*16 floats
  float*  Sp  = (float*)(ws + 4096 + 2048);       // 32*16 floats
  float*  ctxp = (float*)(ws + 8192);             // 32*16*1024 floats = 2 MB

  attn_prep_u<<<dim3(HDIM / 256), dim3(256), 0, stream>>>(Wattn, Wv, ubf);
  attn_partial<<<dim3(NCHUNK, BDIM), dim3(256), 0, stream>>>(enc, ubf, Mp, Sp, ctxp);
  attn_reduce<<<dim3(BDIM), dim3(256), 0, stream>>>(Mp, Sp, ctxp, out);
}
